// CrossAttentionRoPE_64020782514496
// MI455X (gfx1250) — matrix operand order repaired and checked by reading, not compiled, not run on hardware
//
#include <hip/hip_runtime.h>

typedef __attribute__((ext_vector_type(16))) __bf16 v16bf;
typedef __attribute__((ext_vector_type(8)))  float  v8f;

#define B_  8
#define NQ_ 1024
#define NP_ 2048
#define C_  1024
#define H_  16
#define D_  64

// ---------- helpers ----------
__device__ __forceinline__ __bf16 f2bf(float f) {
  unsigned u = __builtin_bit_cast(unsigned, f);
  unsigned r = u + 0x7FFFu + ((u >> 16) & 1u);          // round-to-nearest-even
  unsigned short h = (unsigned short)(r >> 16);
  return __builtin_bit_cast(__bf16, h);
}

// 16 contiguous bf16 from p0 (8) and p1 (8) -> one v16bf fragment
__device__ __forceinline__ v16bf ld16(const __bf16* p0, const __bf16* p1) {
  union { v16bf v; uint4 u[2]; } c;
  c.u[0] = *(const uint4*)p0;
  c.u[1] = *(const uint4*)p1;
  return c.v;
}

__device__ __forceinline__ v8f wmma_bf16(v16bf a, v16bf b, v8f c) {
  return __builtin_amdgcn_wmma_f32_16x16x32_bf16(false, a, false, b, (short)0, c,
                                                 false, false);
}

// ---------- f32 -> bf16 convert ----------
__global__ void cvt_f32_bf16(const float* __restrict__ in, __bf16* __restrict__ out, int n) {
  int i = blockIdx.x * blockDim.x + threadIdx.x;
  if (i < n) out[i] = f2bf(in[i]);
}

// ---------- RoPE on bf16 K, layout (B, NP, H, D) ----------
__global__ void rope_k(__bf16* __restrict__ K, const float* __restrict__ F) {
  int tid = blockIdx.x * blockDim.x + threadIdx.x;
  int total = B_ * NP_ * H_ * (D_ / 2);
  if (tid >= total) return;
  int i = tid & 31;              // D/2 = 32
  int h = (tid >> 5) & (H_ - 1);
  int s = (tid >> 9) & (NP_ - 1);
  int b = tid >> 20;             // 9 + 11
  size_t base = (((size_t)b * NP_ + s) * H_ + h) * D_ + 2 * i;
  float kr = (float)K[base];
  float ki = (float)K[base + 1];
  float c = F[((size_t)s * 32 + i) * 2 + 0];
  float sn = F[((size_t)s * 32 + i) * 2 + 1];
  K[base]     = f2bf(kr * c - ki * sn);
  K[base + 1] = f2bf(kr * sn + ki * c);
}

// ---------- generic NT GEMM: Out[m][n] = sum_k A[m][k] * W[n][k] + bias[n] ----------
// Wave tile: 32(M) x 64(N); k-loop unrolled x2 with ping-pong fragment sets
// (no register rotation copies, no WMMA->VALU WAR hazards in steady state).
// MODE 0: f32 out, natural (m*N + n)
// MODE 1: bf16 out, natural
// MODE 2: bf16 out, transposed per batch: (b*N + n)*NPdim + s  with m = b*NPdim + s
template <int MODE>
__global__ __launch_bounds__(128) void gemm_bf16_nt(
    const __bf16* __restrict__ A, const __bf16* __restrict__ W,
    const float* __restrict__ bias, float* __restrict__ outF,
    __bf16* __restrict__ outB, int N, int K, int NPdim) {
  int wave = threadIdx.x >> 5;
  int lane = threadIdx.x & 31;
  int hl = lane >> 4;
  int l16 = lane & 15;
  int m0 = blockIdx.x * 128 + wave * 32;
  int n0 = blockIdx.y * 64;

  const __bf16* arow0 = A + (size_t)(m0 + l16) * K;
  const __bf16* arow1 = A + (size_t)(m0 + 16 + l16) * K;
  const __bf16* wr0 = W + (size_t)(n0 +  0 + l16) * K + hl * 16;
  const __bf16* wr1 = W + (size_t)(n0 + 16 + l16) * K + hl * 16;
  const __bf16* wr2 = W + (size_t)(n0 + 32 + l16) * K + hl * 16;
  const __bf16* wr3 = W + (size_t)(n0 + 48 + l16) * K + hl * 16;

  v8f acc[8] = {};

  // stage A fragments (k = 0)
  v16bf aA0 = ld16(arow0 + hl * 16, arow0 + hl * 16 + 8);
  v16bf aA1 = ld16(arow1 + hl * 16, arow1 + hl * 16 + 8);
  v16bf bA0 = ld16(wr0, wr0 + 8);
  v16bf bA1 = ld16(wr1, wr1 + 8);
  v16bf bA2 = ld16(wr2, wr2 + 8);
  v16bf bA3 = ld16(wr3, wr3 + 8);

  // K is a multiple of 64 for every call in this problem (K = 1024)
  for (int k0 = 0; k0 < K; k0 += 64) {
    int kB = k0 + 32;
    // stage B loads fly over stage A WMMAs
    v16bf aB0 = ld16(arow0 + kB + hl * 16, arow0 + kB + hl * 16 + 8);
    v16bf aB1 = ld16(arow1 + kB + hl * 16, arow1 + kB + hl * 16 + 8);
    v16bf bB0 = ld16(wr0 + kB, wr0 + kB + 8);
    v16bf bB1 = ld16(wr1 + kB, wr1 + kB + 8);
    v16bf bB2 = ld16(wr2 + kB, wr2 + kB + 8);
    v16bf bB3 = ld16(wr3 + kB, wr3 + kB + 8);
    if (k0 + 128 < K) {
      __builtin_prefetch(arow0 + k0 + 128, 0, 3);
      __builtin_prefetch(arow1 + k0 + 128, 0, 3);
    }

    acc[0] = wmma_bf16(aA0, bA0, acc[0]);
    acc[1] = wmma_bf16(aA0, bA1, acc[1]);
    acc[2] = wmma_bf16(aA0, bA2, acc[2]);
    acc[3] = wmma_bf16(aA0, bA3, acc[3]);
    acc[4] = wmma_bf16(aA1, bA0, acc[4]);
    acc[5] = wmma_bf16(aA1, bA1, acc[5]);
    acc[6] = wmma_bf16(aA1, bA2, acc[6]);
    acc[7] = wmma_bf16(aA1, bA3, acc[7]);

    int kA = k0 + 64;
    if (kA < K) {
      // reload stage A for the next iteration; flies over stage B WMMAs
      aA0 = ld16(arow0 + kA + hl * 16, arow0 + kA + hl * 16 + 8);
      aA1 = ld16(arow1 + kA + hl * 16, arow1 + kA + hl * 16 + 8);
      bA0 = ld16(wr0 + kA, wr0 + kA + 8);
      bA1 = ld16(wr1 + kA, wr1 + kA + 8);
      bA2 = ld16(wr2 + kA, wr2 + kA + 8);
      bA3 = ld16(wr3 + kA, wr3 + kA + 8);
    }

    acc[0] = wmma_bf16(aB0, bB0, acc[0]);
    acc[1] = wmma_bf16(aB0, bB1, acc[1]);
    acc[2] = wmma_bf16(aB0, bB2, acc[2]);
    acc[3] = wmma_bf16(aB0, bB3, acc[3]);
    acc[4] = wmma_bf16(aB1, bB0, acc[4]);
    acc[5] = wmma_bf16(aB1, bB1, acc[5]);
    acc[6] = wmma_bf16(aB1, bB2, acc[6]);
    acc[7] = wmma_bf16(aB1, bB3, acc[7]);
  }

#pragma unroll
  for (int half_m = 0; half_m < 2; half_m++) {
#pragma unroll
    for (int j = 0; j < 4; j++) {
      int col = n0 + 16 * j + l16;
      float bvv = bias[col];
#pragma unroll
      for (int v = 0; v < 8; v++) {
        int m = m0 + half_m * 16 + v + hl * 8;
        float val = acc[half_m * 4 + j][v] + bvv;
        if constexpr (MODE == 0) {
          outF[(size_t)m * N + col] = val;
        } else if constexpr (MODE == 1) {
          outB[(size_t)m * N + col] = f2bf(val);
        } else {
          int bb = m / NPdim, ss = m % NPdim;
          outB[((size_t)bb * N + col) * NPdim + ss] = f2bf(val);
        }
      }
    }
  }
}

// ---------- flash attention ----------
// Q: (B,NQ,H,D) bf16, K: (B,NP,H,D) bf16 (RoPE applied), Vt: (B,H,D,NP) bf16
// Oa: (B,NQ,C) bf16
__global__ __launch_bounds__(128) void attn_kernel(
    const __bf16* __restrict__ Q, const __bf16* __restrict__ K,
    const __bf16* __restrict__ Vt, __bf16* __restrict__ Oa) {
  __shared__ __bf16 ldsP[4][16 * 64];

  int wave = threadIdx.x >> 5;
  int lane = threadIdx.x & 31;
  int hl = lane >> 4;
  int l16 = lane & 15;

  int blk = blockIdx.x;                 // B*H*(NQ/64) = 2048
  int qt = blk & 15;                    // NQ/64 = 16
  int h = (blk >> 4) & (H_ - 1);
  int b = blk >> 8;
  int q0 = qt * 64 + wave * 16;

  // Q A-fragments for the whole D=64 (two k-chunks of 32)
  const __bf16* qrow = Q + (((size_t)b * NQ_ + q0 + l16) * H_ + h) * D_;
  v16bf qa0 = ld16(qrow + hl * 16,      qrow + hl * 16 + 8);
  v16bf qa1 = ld16(qrow + 32 + hl * 16, qrow + 32 + hl * 16 + 8);

  // V fragment base pointers (transposed layout: contiguous along keys)
  const __bf16* vbase0 = Vt + ((size_t)b * C_ + h * D_ +  0 + l16) * NP_;
  const __bf16* vbase1 = Vt + ((size_t)b * C_ + h * D_ + 16 + l16) * NP_;
  const __bf16* vbase2 = Vt + ((size_t)b * C_ + h * D_ + 32 + l16) * NP_;
  const __bf16* vbase3 = Vt + ((size_t)b * C_ + h * D_ + 48 + l16) * NP_;

  v8f o[4] = {};
  float rmax[8], rsum[8];
#pragma unroll
  for (int v = 0; v < 8; v++) { rmax[v] = -1e30f; rsum[v] = 0.0f; }
  const float scale = 0.125f;           // 1/sqrt(64)

  __bf16* lp = &ldsP[wave][0];

  for (int kt = 0; kt < NP_; kt += 64) {
    // ---- load all eight K B-fragments for this 64-key tile ----
    v16bf kb0[4], kb1[4];
#pragma unroll
    for (int j = 0; j < 4; j++) {
      const __bf16* krow = K + (((size_t)b * NP_ + kt + 16 * j + l16) * H_ + h) * D_;
      kb0[j] = ld16(krow + hl * 16,      krow + hl * 16 + 8);
      kb1[j] = ld16(krow + 32 + hl * 16, krow + 32 + hl * 16 + 8);
    }
    if (kt + 64 < NP_) {
      __builtin_prefetch(K + (((size_t)b * NP_ + kt + 64 + l16) * H_ + h) * D_, 0, 3);
      __builtin_prefetch(K + (((size_t)b * NP_ + kt + 96 + l16) * H_ + h) * D_, 0, 3);
    }

    // ---- S = Q * K^T (16x64 score tile) ----
    v8f s[4] = {};
#pragma unroll
    for (int j = 0; j < 4; j++) {
      s[j] = wmma_bf16(qa0, kb0[j], s[j]);
      s[j] = wmma_bf16(qa1, kb1[j], s[j]);
    }

    // ---- issue V loads now; they fly during the softmax VALU work ----
    v16bf vb0[4], vb1[4];
#pragma unroll
    for (int j = 0; j < 4; j++) {
      const __bf16* vrow = (j == 0 ? vbase0 : j == 1 ? vbase1 : j == 2 ? vbase2 : vbase3) + kt;
      vb0[j] = ld16(vrow + hl * 16,      vrow + hl * 16 + 8);
      vb1[j] = ld16(vrow + 32 + hl * 16, vrow + 32 + hl * 16 + 8);
    }

    // ---- online softmax (rows live in one VGPR across a half-wave) ----
    float sc[8];
#pragma unroll
    for (int v = 0; v < 8; v++) {
#pragma unroll
      for (int j = 0; j < 4; j++) s[j][v] *= scale;
      float mv = fmaxf(fmaxf(s[0][v], s[1][v]), fmaxf(s[2][v], s[3][v]));
      mv = fmaxf(mv, __shfl_xor(mv, 1, 32));
      mv = fmaxf(mv, __shfl_xor(mv, 2, 32));
      mv = fmaxf(mv, __shfl_xor(mv, 4, 32));
      mv = fmaxf(mv, __shfl_xor(mv, 8, 32));
      float newm = fmaxf(rmax[v], mv);
      float so = __expf(rmax[v] - newm);
      rmax[v] = newm;
      float part = 0.0f;
#pragma unroll
      for (int j = 0; j < 4; j++) {
        float p = __expf(s[j][v] - newm);
        s[j][v] = p;
        part += p;
      }
      part += __shfl_xor(part, 1, 32);
      part += __shfl_xor(part, 2, 32);
      part += __shfl_xor(part, 4, 32);
      part += __shfl_xor(part, 8, 32);
      rsum[v] = rsum[v] * so + part;
      sc[v] = so;
    }
    // rescale existing accumulators
#pragma unroll
    for (int j = 0; j < 4; j++)
#pragma unroll
      for (int v = 0; v < 8; v++) o[j][v] *= sc[v];

    // ---- stage P (bf16) through LDS: C/D layout -> A layout ----
#pragma unroll
    for (int j = 0; j < 4; j++)
#pragma unroll
      for (int v = 0; v < 8; v++)
        lp[(v + hl * 8) * 64 + 16 * j + l16] = f2bf(s[j][v]);
    // same-wave LDS ops are in order: RAW is safe without a barrier
    const __bf16* pr = lp + l16 * 64;
    v16bf pa0 = ld16(pr + hl * 16,      pr + hl * 16 + 8);
    v16bf pa1 = ld16(pr + 32 + hl * 16, pr + 32 + hl * 16 + 8);

    // ---- O += P * V ----
#pragma unroll
    for (int j = 0; j < 4; j++) {
      o[j] = wmma_bf16(pa0, vb0[j], o[j]);
      o[j] = wmma_bf16(pa1, vb1[j], o[j]);
    }
  }

  // ---- normalize + store to (B,NQ,C) bf16 ----
  float inv[8];
#pragma unroll
  for (int v = 0; v < 8; v++) inv[v] = 1.0f / rsum[v];
#pragma unroll
  for (int j = 0; j < 4; j++) {
    int col = h * D_ + 16 * j + l16;
#pragma unroll
    for (int v = 0; v < 8; v++) {
      int qr = q0 + v + hl * 8;
      Oa[((size_t)b * NQ_ + qr) * C_ + col] = f2bf(o[j][v] * inv[v]);
    }
  }
}

// ---------- host launch ----------
extern "C" void kernel_launch(void* const* d_in, const int* in_sizes, int n_in,
                              void* d_out, int out_size, void* d_ws, size_t ws_size,
                              hipStream_t stream) {
  const float* x    = (const float*)d_in[0];
  const float* ctx  = (const float*)d_in[1];
  const float* fcis = (const float*)d_in[2];
  const float* wq   = (const float*)d_in[3];
  const float* bq   = (const float*)d_in[4];
  const float* wk   = (const float*)d_in[5];
  const float* bk   = (const float*)d_in[6];
  const float* wv   = (const float*)d_in[7];
  const float* bv   = (const float*)d_in[8];
  const float* wo   = (const float*)d_in[9];
  const float* bo   = (const float*)d_in[10];
  float* out = (float*)d_out;

  __bf16* ws = (__bf16*)d_ws;
  const size_t nX  = (size_t)B_ * NQ_ * C_;     // 8388608
  const size_t nCx = (size_t)B_ * NP_ * C_;     // 16777216
  const size_t nW  = (size_t)C_ * C_;           // 1048576
  __bf16* xb   = ws;
  __bf16* ctxb = xb + nX;
  __bf16* wqb  = ctxb + nCx;
  __bf16* wkb  = wqb + nW;
  __bf16* wvb  = wkb + nW;
  __bf16* wob  = wvb + nW;
  __bf16* Qb   = wob + nW;        // (B,NQ,H,D)
  __bf16* Kb   = Qb + nX;         // (B,NP,H,D)
  __bf16* Vt   = Kb + nCx;        // (B,H,D,NP)
  __bf16* At   = Vt + nCx;        // (B,NQ,C)

  // convert to bf16
  cvt_f32_bf16<<<(int)((nX  + 255) / 256), 256, 0, stream>>>(x,   xb,   (int)nX);
  cvt_f32_bf16<<<(int)((nCx + 255) / 256), 256, 0, stream>>>(ctx, ctxb, (int)nCx);
  cvt_f32_bf16<<<(int)((nW  + 255) / 256), 256, 0, stream>>>(wq, wqb, (int)nW);
  cvt_f32_bf16<<<(int)((nW  + 255) / 256), 256, 0, stream>>>(wk, wkb, (int)nW);
  cvt_f32_bf16<<<(int)((nW  + 255) / 256), 256, 0, stream>>>(wv, wvb, (int)nW);
  cvt_f32_bf16<<<(int)((nW  + 255) / 256), 256, 0, stream>>>(wo, wob, (int)nW);

  // projections (wave tile 32x64, block tile 128x64)
  dim3 blk(128);
  dim3 gq(B_ * NQ_ / 128, C_ / 64);   // 64 x 16
  dim3 gk(B_ * NP_ / 128, C_ / 64);   // 128 x 16
  gemm_bf16_nt<1><<<gq, blk, 0, stream>>>(xb,   wqb, bq, nullptr, Qb, C_, C_, NP_);
  gemm_bf16_nt<1><<<gk, blk, 0, stream>>>(ctxb, wkb, bk, nullptr, Kb, C_, C_, NP_);
  gemm_bf16_nt<2><<<gk, blk, 0, stream>>>(ctxb, wvb, bv, nullptr, Vt, C_, C_, NP_);

  // RoPE on K
  int nrope = B_ * NP_ * H_ * (D_ / 2);
  rope_k<<<(nrope + 255) / 256, 256, 0, stream>>>(Kb, fcis);

  // attention
  attn_kernel<<<B_ * H_ * (NQ_ / 64), 128, 0, stream>>>(Qb, Kb, Vt, At);

  // output projection (f32 + bias) -> d_out
  gemm_bf16_nt<0><<<gq, blk, 0, stream>>>(At, wob, bo, out, nullptr, C_, C_, NP_);
}
